// EncoderDecoder_74680891343643
// MI455X (gfx1250) — matrix operand order repaired and checked by reading, not compiled, not run on hardware
//
#include <hip/hip_runtime.h>
#include <hip/hip_bf16.h>
#include <cstddef>

// ---------------- model dimensions ----------------
#define BB 32
#define SS 400
#define EE 256
#define HH 512
#define AA 64
#define TT 20
#define ENUM 50000
#define VOC 50100
#define G3H 1536            // 3*H
#define DIN 896             // E + H + 2A

typedef _Float16 half_t;
typedef __attribute__((ext_vector_type(16))) _Float16 v16h;
typedef __attribute__((ext_vector_type(8)))  _Float16 v8h;
typedef __attribute__((ext_vector_type(8)))  float    v8f;

static __device__ __forceinline__ v8f wmma_f16(v16h a, v16h b, v8f c) {
  // v_wmma_f32_16x16x32_f16 : D = A(16x32) * B(32x16) + C(16x16 f32)
  return __builtin_amdgcn_wmma_f32_16x16x32_f16(false, a, false, b, (short)0, c, false, false);
}

static __device__ __forceinline__ float sigm(float x) { return 1.f / (1.f + expf(-x)); }

// Build A fragment (16x32 f16) for row block m0, k block k0.
// Lanes 0-15: M=lane, elems 0..15 = K k0..k0+15
// Lanes 16-31: M=lane-16, elems 0..15 = K k0+16..k0+31
static __device__ __forceinline__ v16h load_fragA(const half_t* Ap, int k0) {
  v8h lo = *(const v8h*)(Ap + k0);
  v8h hi = *(const v8h*)(Ap + k0 + 8);
  v16h a;
#pragma unroll
  for (int q = 0; q < 8; ++q) { a[q] = lo[q]; a[8 + q] = hi[q]; }
  return a;
}

// ---------------- generic converters / gathers ----------------
__global__ void ed_cvt_f16(const float* __restrict__ s, half_t* __restrict__ d, int n) {
  int i = blockIdx.x * blockDim.x + threadIdx.x;
  if (i < n) d[i] = (half_t)s[i];
}

__global__ void ed_zero_acc(float* acc) {
  if (threadIdx.x < 2) acc[threadIdx.x] = 0.f;
}

__global__ void ed_write_scalars(const float* acc, float* out) {
  if (threadIdx.x == 0) { out[0] = acc[0]; out[1] = acc[1]; }
}

__global__ void ed_gather_embed(const int* __restrict__ idx, const float* __restrict__ embW,
                                half_t* __restrict__ x0) {
  int i = blockIdx.x * blockDim.x + threadIdx.x;
  if (i >= BB * SS * EE) return;
  int e = i & (EE - 1);
  int bs = i >> 8;                       // EE = 256
  x0[i] = (half_t)embW[(size_t)idx[bs] * EE + e];
}

__global__ void ed_gather_up(const int* __restrict__ user, const int* __restrict__ prod,
                             const float* __restrict__ userW, const float* __restrict__ prodW,
                             float* __restrict__ up, half_t* __restrict__ up16) {
  int i = blockIdx.x * blockDim.x + threadIdx.x;
  if (i >= BB * 2 * AA) return;
  int b = i >> 7, j = i & 127;
  float v = (j < AA) ? userW[(size_t)user[b] * AA + j] : prodW[(size_t)prod[b] * AA + (j - AA)];
  up[i] = v; up16[i] = (half_t)v;
}

// ---------------- generic WMMA GEMM:  C[M,N] = act(A[M,K] * W[N,K]^T + bias) ----------------
// block = 256 threads (8 waves), tile 32(M) x 64(N); act: 0=none 1=tanh 3=leaky(0.01)
__global__ __launch_bounds__(256) void ed_wmma_gemm(
    const half_t* __restrict__ A, const half_t* __restrict__ W,
    const float* __restrict__ bias, float* __restrict__ C, half_t* __restrict__ C16,
    int M, int N, int K, int act) {
  int wave = threadIdx.x >> 5;
  int lane = threadIdx.x & 31;
  int mt = wave >> 2, nt = wave & 3;
  int m0 = blockIdx.y * 32 + mt * 16;
  int n0 = blockIdx.x * 64 + nt * 16;
  if (n0 >= N || m0 >= M) return;      // wave-uniform skip (EXEC stays all-ones per wave)
  int lrow = lane & 15, lhi = lane >> 4;
  const half_t* Ap = A + (size_t)(m0 + lrow) * K + (lhi << 4);
  const half_t* Wp = W + (size_t)(n0 + lrow) * K + (lhi << 4);
  v8f acc = {0.f, 0.f, 0.f, 0.f, 0.f, 0.f, 0.f, 0.f};
  for (int k0 = 0; k0 < K; k0 += 32) {
    v16h a = load_fragA(Ap, k0);
    v16h b = *(const v16h*)(Wp + k0);  // lanes 0-15: K k0..k0+15 of col n0+lane; 16-31: +16
    acc = wmma_f16(a, b, acc);
  }
  int n = n0 + lrow;
  float bv = bias ? bias[n] : 0.f;
  int mb = m0 + (lhi << 3);
#pragma unroll
  for (int r = 0; r < 8; ++r) {
    float v = acc[r] + bv;
    if (act == 1) v = tanhf(v);
    else if (act == 3) v = (v > 0.f) ? v : 0.01f * v;
    size_t o = (size_t)(mb + r) * N + n;
    C[o] = v;
    if (C16) C16[o] = (half_t)v;
  }
}

// ---------------- persistent GRU scan (one workgroup, h kept in LDS) ----------------
// gi: [B,S,3H] precomputed input gates (f32).  Wh: [3H,H] f16.  bh: [3H] f32.
// out16: [B,S,2H] f16, written at feature offset out_off.
// hfin/hfin16: final hidden, row stride 2H (pre-offset by caller for direction slice).
__global__ __launch_bounds__(1024) void ed_gru_scan(
    const float* __restrict__ gi, const half_t* __restrict__ Wh, const float* __restrict__ bh,
    half_t* __restrict__ out16, int out_off,
    float* __restrict__ hfin, half_t* __restrict__ hfin16, int reverse) {
  extern __shared__ char ed_smem[];
  half_t* h16 = (half_t*)ed_smem;                       // [32,512] f16 = 32 KB
  float* gh = (float*)(ed_smem + BB * HH * sizeof(half_t));  // [32,1536] f32 = 192 KB
  int tid = threadIdx.x;
  int wave = tid >> 5, lane = tid & 31;
  int lrow = lane & 15, lhi = lane >> 4;
  for (int i = tid; i < BB * HH; i += 1024) h16[i] = (half_t)0.f;
  __syncthreads();
  for (int idx = 0; idx < SS; ++idx) {
    int t = reverse ? (SS - 1 - idx) : idx;
    // --- gh = h @ Wh^T + bh  (32x1536, K=512): 192 tiles over 32 waves ---
    for (int tile = wave; tile < 192; tile += 32) {
      int m0 = (tile & 1) * 16;
      int n0 = (tile >> 1) * 16;
      const half_t* Ap = h16 + (m0 + lrow) * HH + (lhi << 4);
      const half_t* Wp = Wh + (size_t)(n0 + lrow) * HH + (lhi << 4);
      v8f acc = {0.f, 0.f, 0.f, 0.f, 0.f, 0.f, 0.f, 0.f};
#pragma unroll 4
      for (int k0 = 0; k0 < HH; k0 += 32) {
        v16h a = load_fragA(Ap, k0);
        v16h b = *(const v16h*)(Wp + k0);
        acc = wmma_f16(a, b, acc);
      }
      int n = n0 + lrow;
      float bv = bh[n];
      int mb = m0 + (lhi << 3);
#pragma unroll
      for (int r = 0; r < 8; ++r) gh[(mb + r) * G3H + n] = acc[r] + bv;
    }
    __syncthreads();
    // --- gate nonlinearity, update h ---
    for (int i = tid; i < BB * HH; i += 1024) {
      int b = i >> 9, hh = i & (HH - 1);
      const float* gib = gi + ((size_t)b * SS + t) * G3H;
      const float* ghb = gh + b * G3H;
      float r = sigm(gib[hh] + ghb[hh]);
      float z = sigm(gib[HH + hh] + ghb[HH + hh]);
      float nn = tanhf(gib[2 * HH + hh] + r * ghb[2 * HH + hh]);
      float hn = (1.f - z) * nn + z * (float)h16[i];
      h16[i] = (half_t)hn;
      out16[((size_t)b * SS + t) * (2 * HH) + out_off + hh] = (half_t)hn;
    }
    __syncthreads();
  }
  for (int i = tid; i < BB * HH; i += 1024) {
    int b = i >> 9, hh = i & (HH - 1);
    float v = (float)h16[i];
    hfin[(size_t)b * (2 * HH) + hh] = v;
    hfin16[(size_t)b * (2 * HH) + hh] = (half_t)v;
  }
}

// ---------------- encoder mixing ----------------
__global__ void ed_mix_hidden(const float* __restrict__ eg, const float* __restrict__ tf,
                              const float* __restrict__ af, float* __restrict__ hdec,
                              half_t* __restrict__ hdec16, float* __restrict__ ctxh) {
  int i = blockIdx.x * blockDim.x + threadIdx.x;
  if (i >= 2 * BB * HH) return;
  int l = i / (BB * HH);
  int r = i - l * (BB * HH);
  int b = r >> 9, h = r & (HH - 1);
  float e = eg[b];
  float v = e * tf[(size_t)(l * BB + b) * HH + h] + (1.f - e) * af[(size_t)b * 1024 + l * HH + h];
  hdec[i] = v; hdec16[i] = (half_t)v;
  if (l == 1) ctxh[b * HH + h] = v;
}

// ---------------- generic concat-dot gate: g[b]=sigmoid(sum_j W.seg_j + bias) ----------------
__global__ void ed_concat_gate(const float* __restrict__ W, const float* __restrict__ bias,
                               const float* s0, int l0, int st0,
                               const float* s1, int l1, int st1,
                               const float* s2, int l2, int st2,
                               const float* s3, int l3, int st3,
                               float* __restrict__ g, float* acc, float scale) {
  int b = blockIdx.x, lane = threadIdx.x;
  float sum = 0.f; int base = 0;
  if (s0) { for (int i = lane; i < l0; i += 32) sum += W[base + i] * s0[(size_t)b * st0 + i]; base += l0; }
  if (s1) { for (int i = lane; i < l1; i += 32) sum += W[base + i] * s1[(size_t)b * st1 + i]; base += l1; }
  if (s2) { for (int i = lane; i < l2; i += 32) sum += W[base + i] * s2[(size_t)b * st2 + i]; base += l2; }
  if (s3) { for (int i = lane; i < l3; i += 32) sum += W[base + i] * s3[(size_t)b * st3 + i]; base += l3; }
  for (int off = 16; off; off >>= 1) sum += __shfl_down(sum, off);
  if (lane == 0) {
    float gv = sigm(sum + bias[0]);
    g[b] = gv;
    if (acc) atomicAdd(acc, gv * scale);
  }
}

// ---------------- decoder helpers ----------------
__global__ void ed_build_dec_x(const int* __restrict__ trg, const float* __restrict__ embW,
                               const float* __restrict__ ctxh, const float* __restrict__ up,
                               half_t* __restrict__ x16, float* __restrict__ prevf, int t) {
  int i = blockIdx.x * blockDim.x + threadIdx.x;
  if (i >= BB * DIN) return;
  int b = i / DIN, j = i - b * DIN;
  float v;
  if (j < EE) {
    int tok = (t == 0) ? 1 : trg[b * TT + (t - 1)];
    v = embW[(size_t)tok * EE + j];
    prevf[b * EE + j] = v;
  } else if (j < EE + HH) {
    v = ctxh[b * HH + (j - EE)];
  } else if (j < EE + HH + AA) {
    v = up[b * 2 * AA + (j - EE - HH)];
  } else {
    v = up[b * 2 * AA + AA + (j - EE - HH - AA)];
  }
  x16[i] = (half_t)v;
}

__global__ void ed_gru_gate(const float* __restrict__ gi, const float* __restrict__ gh,
                            float* __restrict__ h, half_t* __restrict__ h16) {
  int i = blockIdx.x * blockDim.x + threadIdx.x;
  if (i >= BB * HH) return;
  int b = i >> 9, hh = i & (HH - 1);
  float r = sigm(gi[b * G3H + hh] + gh[b * G3H + hh]);
  float z = sigm(gi[b * G3H + HH + hh] + gh[b * G3H + HH + hh]);
  float n = tanhf(gi[b * G3H + 2 * HH + hh] + r * gh[b * G3H + 2 * HH + hh]);
  float hn = (1.f - z) * n + z * h[i];
  h[i] = hn; h16[i] = (half_t)hn;
}

__global__ void ed_attn_scores(const float* __restrict__ qp, const float* __restrict__ pk,
                               const float* __restrict__ energy, const int* __restrict__ mask,
                               float* __restrict__ scores) {
  int gw = (blockIdx.x * blockDim.x + threadIdx.x) >> 5;
  int lane = threadIdx.x & 31;
  if (gw >= BB * SS) return;
  int b = gw / SS, s = gw - b * SS;
  const float* pks = pk + ((size_t)b * SS + s) * HH;
  const float* qpb = qp + b * HH;
  float sum = 0.f;
  for (int h = lane; h < HH; h += 32) sum += tanhf(qpb[h] + pks[h]) * energy[h];
  for (int off = 16; off; off >>= 1) sum += __shfl_down(sum, off);
  if (lane == 0) scores[gw] = mask[gw] ? sum : -1e30f;
}

__global__ void ed_softmax_rows(const float* __restrict__ scores, float* __restrict__ alphas) {
  int b = blockIdx.x, tid = threadIdx.x;   // 256 threads, S=400
  __shared__ float red[256];
  float m = -1e30f;
  for (int s = tid; s < SS; s += 256) m = fmaxf(m, scores[b * SS + s]);
  red[tid] = m; __syncthreads();
  for (int k = 128; k; k >>= 1) { if (tid < k) red[tid] = fmaxf(red[tid], red[tid + k]); __syncthreads(); }
  m = red[0]; __syncthreads();
  float sum = 0.f;
  for (int s = tid; s < SS; s += 256) sum += expf(scores[b * SS + s] - m);
  red[tid] = sum; __syncthreads();
  for (int k = 128; k; k >>= 1) { if (tid < k) red[tid] += red[tid + k]; __syncthreads(); }
  sum = red[0];
  for (int s = tid; s < SS; s += 256) alphas[b * SS + s] = expf(scores[b * SS + s] - m) / sum;
}

__global__ void ed_ctx_text(const float* __restrict__ alphas, const half_t* __restrict__ eh,
                            float* __restrict__ ctxt, half_t* __restrict__ ctxt16) {
  int i = blockIdx.x * blockDim.x + threadIdx.x;
  if (i >= BB * 2 * HH) return;
  int b = i >> 10, d = i & 1023;
  const half_t* ehb = eh + (size_t)b * SS * (2 * HH) + d;
  const float* ab = alphas + b * SS;
  float sum = 0.f;
  for (int s = 0; s < SS; ++s) sum += ab[s] * (float)ehb[(size_t)s * (2 * HH)];
  ctxt[i] = sum; ctxt16[i] = (half_t)sum;
}

__global__ void ed_attr_attn(const float* __restrict__ qpa, const float* __restrict__ pka,
                             const float* __restrict__ energyA, const float* __restrict__ up,
                             float* __restrict__ ctxattr, half_t* __restrict__ ctxattr16) {
  int b = blockIdx.x, tid = threadIdx.x;   // 64 threads = 2 waves
  int k = tid >> 5, lane = tid & 31;
  __shared__ float s2[2];
  float sum = 0.f;
  for (int h = lane; h < HH; h += 32)
    sum += tanhf(qpa[b * HH + h] + pka[(size_t)(b * 2 + k) * HH + h]) * energyA[h];
  for (int off = 16; off; off >>= 1) sum += __shfl_down(sum, off);
  if (lane == 0) s2[k] = sum;
  __syncthreads();
  float m = fmaxf(s2[0], s2[1]);
  float e0 = expf(s2[0] - m), e1 = expf(s2[1] - m);
  float a0 = e0 / (e0 + e1), a1 = e1 / (e0 + e1);
  if (tid < AA) {
    float v = a0 * up[b * 2 * AA + tid] + a1 * up[b * 2 * AA + AA + tid];
    ctxattr[b * AA + tid] = v; ctxattr16[b * AA + tid] = (half_t)v;
  }
}

__global__ void ed_mix_context(const float* __restrict__ cg, const float* __restrict__ tc,
                               const float* __restrict__ ac, const float* __restrict__ h1,
                               float* __restrict__ ctx, half_t* __restrict__ xq16) {
  int i = blockIdx.x * blockDim.x + threadIdx.x;
  if (i >= BB * HH) return;
  int b = i >> 9, hh = i & (HH - 1);
  float g = cg[b];
  float c = g * tc[i] + (1.f - g) * ac[i];
  ctx[i] = c;
  xq16[b * 1024 + hh] = (half_t)h1[i];
  xq16[b * 1024 + HH + hh] = (half_t)c;
}

__global__ void ed_gen_softmax_mix(const float* __restrict__ logits, const float* __restrict__ gp,
                                   float* __restrict__ out, int t) {
  int b = blockIdx.x, tid = threadIdx.x;   // 512 threads
  __shared__ float red[512];
  const float* lb = logits + (size_t)b * ENUM;
  float m = -1e30f;
  for (int v = tid; v < ENUM; v += 512) m = fmaxf(m, lb[v]);
  red[tid] = m; __syncthreads();
  for (int k = 256; k; k >>= 1) { if (tid < k) red[tid] = fmaxf(red[tid], red[tid + k]); __syncthreads(); }
  m = red[0]; __syncthreads();
  float sum = 0.f;
  for (int v = tid; v < ENUM; v += 512) sum += expf(lb[v] - m);
  red[tid] = sum; __syncthreads();
  for (int k = 256; k; k >>= 1) { if (tid < k) red[tid] += red[tid + k]; __syncthreads(); }
  sum = red[0];
  float g = gp[b], inv = 1.f / sum;
  float* ob = out + ((size_t)b * TT + t) * VOC;
  for (int v = tid; v < VOC; v += 512) ob[v] = (v < ENUM) ? g * expf(lb[v] - m) * inv : 0.f;
}

__global__ void ed_copy_scatter(const int* __restrict__ src, const float* __restrict__ alphas,
                                const float* __restrict__ gp, float* __restrict__ out, int t) {
  int i = blockIdx.x * blockDim.x + threadIdx.x;
  if (i >= BB * SS) return;
  int b = i / SS;
  atomicAdd(out + ((size_t)b * TT + t) * VOC + src[i], (1.f - gp[b]) * alphas[i]);
}

// =======================================================================
extern "C" void kernel_launch(void* const* d_in, const int* in_sizes, int n_in,
                              void* d_out, int out_size, void* d_ws, size_t ws_size,
                              hipStream_t stream) {
  // ---- inputs ----
  const int* src      = (const int*)d_in[0];
  const int* src_     = (const int*)d_in[1];
  const int* trg_     = (const int*)d_in[2];
  const int* user     = (const int*)d_in[3];
  const int* product  = (const int*)d_in[4];
  const int* src_mask = (const int*)d_in[5];
  const float* embed_W   = (const float*)d_in[7];
  const float* enc_Wi0   = (const float*)d_in[8];
  const float* enc_Wh0   = (const float*)d_in[9];
  const float* enc_bi0   = (const float*)d_in[10];
  const float* enc_bh0   = (const float*)d_in[11];
  const float* enc_Wi1   = (const float*)d_in[12];
  const float* enc_Wh1   = (const float*)d_in[13];
  const float* enc_bi1   = (const float*)d_in[14];
  const float* enc_bh1   = (const float*)d_in[15];
  const float* tf_b      = (const float*)d_in[17];
  const float* user_W    = (const float*)d_in[18];
  const float* prod_W    = (const float*)d_in[19];
  const float* ea_b      = (const float*)d_in[21];
  const float* eg_W      = (const float*)d_in[22];
  const float* eg_b      = (const float*)d_in[23];
  const float* dec_bi0   = (const float*)d_in[26];
  const float* dec_bh0   = (const float*)d_in[27];
  const float* dec_bi1   = (const float*)d_in[30];
  const float* dec_bh1   = (const float*)d_in[31];
  const float* energyT   = (const float*)d_in[34];
  const float* energyA   = (const float*)d_in[37];
  const float* tc_b      = (const float*)d_in[39];
  const float* ac_b      = (const float*)d_in[41];
  const float* cg_W      = (const float*)d_in[42];
  const float* cg_b      = (const float*)d_in[43];
  const float* ch_b      = (const float*)d_in[45];
  const float* gp_W      = (const float*)d_in[46];
  const float* gp_b      = (const float*)d_in[47];
  float* out = (float*)d_out;

  // ---- workspace bump allocator ----
  size_t cur = 0;
  auto alloc = [&](size_t bytes) -> char* {
    char* p = (char*)d_ws + cur;
    cur += (bytes + 255) & ~(size_t)255;
    return p;
  };
  auto allocH = [&](size_t n) { return (half_t*)alloc(n * sizeof(half_t)); };
  auto allocF = [&](size_t n) { return (float*)alloc(n * sizeof(float)); };

  // f16 weight copies
  half_t* wi0  = allocH(2 * G3H * EE);
  half_t* wh0  = allocH(2 * G3H * HH);
  half_t* wi1  = allocH(2 * G3H * 1024);
  half_t* wh1  = allocH(2 * G3H * HH);
  half_t* dwi0 = allocH(G3H * DIN);
  half_t* dwh0 = allocH(G3H * HH);
  half_t* dwi1 = allocH(G3H * HH);
  half_t* dwh1 = allocH(G3H * HH);
  half_t* tfW  = allocH(HH * 1024);
  half_t* akW  = allocH(HH * 1024);
  half_t* aqW  = allocH(HH * HH);
  half_t* arkW = allocH(HH * AA);
  half_t* arqW = allocH(HH * HH);
  half_t* tcW  = allocH(HH * 1024);
  half_t* acW  = allocH(HH * AA);
  half_t* chW  = allocH(HH * 1024);
  half_t* eaW  = allocH(1024 * 128);
  half_t* genW = allocH((size_t)ENUM * HH);
  // activations
  half_t* x0_16   = allocH((size_t)BB * SS * EE);
  float*  gi_big  = allocF((size_t)BB * SS * G3H);
  half_t* out0_16 = allocH((size_t)BB * SS * 2 * HH);
  half_t* out1_16 = allocH((size_t)BB * SS * 2 * HH);   // enc_hidden
  float*  projkey = allocF((size_t)BB * SS * HH);
  float*  encfin  = allocF(2 * BB * 2 * HH);
  half_t* encfin16= allocH(2 * BB * 2 * HH);
  float*  textfin = allocF(2 * BB * HH);
  float*  up      = allocF(BB * 2 * AA);
  half_t* up16    = allocH(BB * 2 * AA);
  float*  attrfin = allocF(BB * 1024);
  float*  pkattr  = allocF(2 * BB * HH);
  float*  hdec    = allocF(2 * BB * HH);
  half_t* hdec16  = allocH(2 * BB * HH);
  float*  ctxh    = allocF(BB * HH);          // recurrent ctx_h (f32 carry)
  half_t* ctxh16  = allocH(BB * HH);
  float*  prevf   = allocF(BB * EE);
  half_t* xdec16  = allocH(BB * DIN);
  float*  gi_d    = allocF(BB * G3H);
  float*  gh_d    = allocF(BB * G3H);
  float*  qproj   = allocF(BB * HH);
  float*  qprojA  = allocF(BB * HH);
  float*  scores  = allocF(BB * SS);
  float*  alphas  = allocF(BB * SS);
  float*  ctxt    = allocF(BB * 2 * HH);
  half_t* ctxt16  = allocH(BB * 2 * HH);
  float*  ctxattr = allocF(BB * AA);
  half_t* ctxattr16 = allocH(BB * AA);
  float*  tc      = allocF(BB * HH);
  float*  ac      = allocF(BB * HH);
  float*  egv     = allocF(BB);
  float*  cgv     = allocF(BB);
  float*  gpv     = allocF(BB);
  float*  context = allocF(BB * HH);
  half_t* xq16    = allocH(BB * 1024);
  float*  logits  = allocF((size_t)BB * ENUM);
  float*  accs    = allocF(2);                // g1, g2

  auto cvt = [&](const float* s, half_t* d, size_t n) {
    ed_cvt_f16<<<dim3((unsigned)((n + 255) / 256)), dim3(256), 0, stream>>>(s, d, (int)n);
  };
  auto gemm = [&](const half_t* A, const half_t* W, const float* bias, float* C,
                  half_t* C16, int M, int N, int K, int act) {
    dim3 g((unsigned)((N + 63) / 64), (unsigned)(M / 32));
    ed_wmma_gemm<<<g, dim3(256), 0, stream>>>(A, W, bias, C, C16, M, N, K, act);
  };
  const size_t scan_lds = (size_t)BB * HH * sizeof(half_t) + (size_t)BB * G3H * sizeof(float);

  // ---- phase 0: zero accumulators, convert weights ----
  ed_zero_acc<<<1, 32, 0, stream>>>(accs);
  cvt(enc_Wi0, wi0, 2 * G3H * EE);
  cvt(enc_Wh0, wh0, 2 * G3H * HH);
  cvt(enc_Wi1, wi1, 2 * G3H * 1024);
  cvt(enc_Wh1, wh1, 2 * G3H * HH);
  cvt((const float*)d_in[24], dwi0, G3H * DIN);
  cvt((const float*)d_in[25], dwh0, G3H * HH);
  cvt((const float*)d_in[28], dwi1, G3H * HH);
  cvt((const float*)d_in[29], dwh1, G3H * HH);
  cvt((const float*)d_in[16], tfW,  HH * 1024);
  cvt((const float*)d_in[32], akW,  HH * 1024);
  cvt((const float*)d_in[33], aqW,  HH * HH);
  cvt((const float*)d_in[35], arkW, HH * AA);
  cvt((const float*)d_in[36], arqW, HH * HH);
  cvt((const float*)d_in[38], tcW,  HH * 1024);
  cvt((const float*)d_in[40], acW,  HH * AA);
  cvt((const float*)d_in[44], chW,  HH * 1024);
  cvt((const float*)d_in[20], eaW,  1024 * 128);
  cvt((const float*)d_in[48], genW, (size_t)ENUM * HH);

  // ---- encoder ----
  ed_gather_embed<<<dim3(BB * SS * EE / 256), dim3(256), 0, stream>>>(src_, embed_W, x0_16);
  ed_gather_up<<<dim3((BB * 2 * AA + 255) / 256), dim3(256), 0, stream>>>(user, product, user_W, prod_W, up, up16);

  const int M_enc = BB * SS;
  // layer 0 fwd
  gemm(x0_16, wi0, enc_bi0, gi_big, nullptr, M_enc, G3H, EE, 0);
  ed_gru_scan<<<1, 1024, (unsigned)scan_lds, stream>>>(gi_big, wh0, enc_bh0, out0_16, 0,
      encfin + 0 * BB * 1024 + 0, encfin16 + 0 * BB * 1024 + 0, 0);
  // layer 0 bwd
  gemm(x0_16, wi0 + (size_t)G3H * EE, enc_bi0 + G3H, gi_big, nullptr, M_enc, G3H, EE, 0);
  ed_gru_scan<<<1, 1024, (unsigned)scan_lds, stream>>>(gi_big, wh0 + (size_t)G3H * HH, enc_bh0 + G3H,
      out0_16, HH, encfin + 0 * BB * 1024 + HH, encfin16 + 0 * BB * 1024 + HH, 1);
  // layer 1 fwd
  gemm(out0_16, wi1, enc_bi1, gi_big, nullptr, M_enc, G3H, 1024, 0);
  ed_gru_scan<<<1, 1024, (unsigned)scan_lds, stream>>>(gi_big, wh1, enc_bh1, out1_16, 0,
      encfin + 1 * BB * 1024 + 0, encfin16 + 1 * BB * 1024 + 0, 0);
  // layer 1 bwd
  gemm(out0_16, wi1 + (size_t)G3H * 1024, enc_bi1 + G3H, gi_big, nullptr, M_enc, G3H, 1024, 0);
  ed_gru_scan<<<1, 1024, (unsigned)scan_lds, stream>>>(gi_big, wh1 + (size_t)G3H * HH, enc_bh1 + G3H,
      out1_16, HH, encfin + 1 * BB * 1024 + HH, encfin16 + 1 * BB * 1024 + HH, 1);

  // text_final = leaky_relu(encoder_final @ tfW^T + b)   [2*B, 512]
  gemm(encfin16, tfW, tf_b, textfin, nullptr, 2 * BB, HH, 1024, 3);
  // attr_final = tanh(up @ eaW^T + b)                    [B, 1024]
  gemm(up16, eaW, ea_b, attrfin, nullptr, BB, 1024, 128, 1);
  // eg gate + g1
  ed_concat_gate<<<dim3(BB), dim3(32), 0, stream>>>(eg_W, eg_b,
      encfin + 1 * BB * 1024, 1024, 1024, up, AA, 2 * AA, up + AA, AA, 2 * AA,
      (const float*)nullptr, 0, 0, egv, accs + 0, 1.f / BB);
  ed_mix_hidden<<<dim3((2 * BB * HH + 255) / 256), dim3(256), 0, stream>>>(egv, textfin, attrfin, hdec, hdec16, ctxh);
  // attention keys
  gemm(out1_16, akW, nullptr, projkey, nullptr, M_enc, HH, 1024, 0);
  gemm(up16, arkW, nullptr, pkattr, nullptr, 2 * BB, HH, AA, 0);

  // ---- decoder: 20 teacher-forced steps ----
  float* h0 = hdec;             half_t* h0_16 = hdec16;
  float* h1 = hdec + BB * HH;   half_t* h1_16 = hdec16 + BB * HH;
  for (int t = 0; t < TT; ++t) {
    ed_build_dec_x<<<dim3((BB * DIN + 255) / 256), dim3(256), 0, stream>>>(trg_, embed_W, ctxh, up, xdec16, prevf, t);
    // GRU layer 0
    gemm(xdec16, dwi0, dec_bi0, gi_d, nullptr, BB, G3H, DIN, 0);
    gemm(h0_16, dwh0, dec_bh0, gh_d, nullptr, BB, G3H, HH, 0);
    ed_gru_gate<<<dim3(BB * HH / 256), dim3(256), 0, stream>>>(gi_d, gh_d, h0, h0_16);
    // GRU layer 1
    gemm(h0_16, dwi1, dec_bi1, gi_d, nullptr, BB, G3H, HH, 0);
    gemm(h1_16, dwh1, dec_bh1, gh_d, nullptr, BB, G3H, HH, 0);
    ed_gru_gate<<<dim3(BB * HH / 256), dim3(256), 0, stream>>>(gi_d, gh_d, h1, h1_16);
    // query projections
    gemm(h1_16, aqW, nullptr, qproj, nullptr, BB, HH, HH, 0);
    gemm(h1_16, arqW, nullptr, qprojA, nullptr, BB, HH, HH, 0);
    // text attention
    ed_attn_scores<<<dim3(BB * SS * 32 / 256), dim3(256), 0, stream>>>(qproj, projkey, energyT, src_mask, scores);
    ed_softmax_rows<<<dim3(BB), dim3(256), 0, stream>>>(scores, alphas);
    ed_ctx_text<<<dim3(BB * 2 * HH / 256), dim3(256), 0, stream>>>(alphas, out1_16, ctxt, ctxt16);
    // attribute attention
    ed_attr_attn<<<dim3(BB), dim3(64), 0, stream>>>(qprojA, pkattr, energyA, up, ctxattr, ctxattr16);
    // contexts
    gemm(ctxt16, tcW, tc_b, tc, nullptr, BB, HH, 1024, 1);
    gemm(ctxattr16, acW, ac_b, ac, nullptr, BB, HH, AA, 1);
    ed_concat_gate<<<dim3(BB), dim3(32), 0, stream>>>(cg_W, cg_b,
        ctxt, 1024, 1024, ctxattr, AA, AA, h1, HH, HH, prevf, EE, EE,
        cgv, (t < 3) ? (accs + 1) : nullptr, 1.f / (BB * 3));
    ed_mix_context<<<dim3(BB * HH / 256), dim3(256), 0, stream>>>(cgv, tc, ac, h1, context, xq16);
    // ctx_h' = tanh([query, context] @ chW^T + b) -> overwrites recurrent carry
    gemm(xq16, chW, ch_b, ctxh, ctxh16, BB, HH, 1024, 1);
    ed_concat_gate<<<dim3(BB), dim3(32), 0, stream>>>(gp_W, gp_b,
        context, HH, HH, h1, HH, HH, prevf, EE, EE, (const float*)nullptr, 0, 0,
        gpv, nullptr, 0.f);
    // generation distribution (big GEMM: 32 x 50000 x 512, genW resident in L2)
    gemm(ctxh16, genW, nullptr, logits, nullptr, BB, ENUM, HH, 0);
    ed_gen_softmax_mix<<<dim3(BB), dim3(512), 0, stream>>>(logits, gpv, out, t);
    ed_copy_scatter<<<dim3(BB * SS / 256), dim3(256), 0, stream>>>(src, alphas, gpv, out, t);
  }
  ed_write_scalars<<<1, 32, 0, stream>>>(accs, out + (size_t)BB * TT * VOC);
}
